// SelfAttention_87514253623329
// MI455X (gfx1250) — matrix operand order repaired and checked by reading, not compiled, not run on hardware
//
#include <hip/hip_runtime.h>

typedef _Float16 half_t;
typedef __attribute__((ext_vector_type(16))) _Float16 v16h;
typedef __attribute__((ext_vector_type(8)))  _Float16 v8h;
typedef __attribute__((ext_vector_type(8)))  float    v8f;

#define BB 4
#define NQd 4096
#define NKd 4096
#define DDd 256

// ---------------------------------------------------------------- conversion
__global__ __launch_bounds__(256)
void cvt_f32_f16_k(const float* __restrict__ in, half_t* __restrict__ out, long n) {
  long i = (long)blockIdx.x * blockDim.x + threadIdx.x;
  long stride = (long)gridDim.x * blockDim.x;
  for (; i < n; i += stride) out[i] = (half_t)in[i];
}

// ---------------------------------------------------------------- GEMM core
// C[m][n] = act( scale * sum_k A[m][k] * B[n][k] + bias[n] )
//   A: f16 (AF32=false) or f32 (AF32=true), row-major, leading dim lda (K-contig)
//   B: f16, [N][K] row-major, leading dim ldb (K-contig)  == torch W / K-matrix / vT
//   OUTMODE: 0 = f16 row-major [m][n] (ldc), 1 = f16 transposed [n][m] (ldc),
//            2 = f32 row-major [m][n] (ldc)
// Block: 256 thr = 8 waves; block tile 128x128; wave tile 32x64 (2x4 WMMA accs).
template<bool AF32, int OUTMODE, bool RELU>
__global__ __launch_bounds__(256)
void gemm_wmma_k(const void* __restrict__ Av, const half_t* __restrict__ Bv,
                 const float* __restrict__ bias, void* __restrict__ Cv,
                 int K, int lda, int ldb, int ldc,
                 long sA, long sB, long sC, float scale) {
  const int lane = threadIdx.x & 31;
  const int wave = threadIdx.x >> 5;
  const int wm = blockIdx.y * 128 + (wave >> 1) * 32;   // wave M origin
  const int wn = blockIdx.x * 128 + (wave & 1) * 64;    // wave N origin
  const long aOff = (long)blockIdx.z * sA;
  const long bOff = (long)blockIdx.z * sB;
  const long cOff = (long)blockIdx.z * sC;

  const int arow0 = wm + (lane & 15);        // A fragment row (M)
  const int kb_a  = (lane >> 4) << 4;        // +0 / +16 (A K sub-offset)
  const int brow0 = wn + (lane & 15);        // B fragment row (N)
  const int kb_b  = (lane >> 4) << 4;        // +0 / +16 (B K sub-offset)

  v8f acc[2][4];
  #pragma unroll
  for (int mt = 0; mt < 2; ++mt)
    #pragma unroll
    for (int nt = 0; nt < 4; ++nt)
      acc[mt][nt] = (v8f){0.f,0.f,0.f,0.f,0.f,0.f,0.f,0.f};

  for (int k = 0; k < K; k += 32) {
    v16h afrag[2];
    #pragma unroll
    for (int mt = 0; mt < 2; ++mt) {
      const long rowOff = aOff + (long)(arow0 + 16 * mt) * lda + k + kb_a;
      if constexpr (AF32) {
        const float* p = (const float*)Av + rowOff;
        v16h r;
        #pragma unroll
        for (int i = 0; i < 8; ++i) { r[i] = (half_t)p[i]; r[i + 8] = (half_t)p[8 + i]; }
        afrag[mt] = r;
      } else {
        const half_t* p = (const half_t*)Av + rowOff;
        v8h lo = *(const v8h*)p;
        v8h hi = *(const v8h*)(p + 8);
        v16h r;
        #pragma unroll
        for (int i = 0; i < 8; ++i) { r[i] = lo[i]; r[i + 8] = hi[i]; }
        afrag[mt] = r;
      }
    }
    v16h bfrag[4];
    #pragma unroll
    for (int nt = 0; nt < 4; ++nt) {
      const half_t* p = Bv + bOff + (long)(brow0 + 16 * nt) * ldb + k + kb_b;
      bfrag[nt] = *(const v16h*)p;           // 32B contiguous
    }
    #pragma unroll
    for (int mt = 0; mt < 2; ++mt)
      #pragma unroll
      for (int nt = 0; nt < 4; ++nt)
        acc[mt][nt] = __builtin_amdgcn_wmma_f32_16x16x32_f16(
            false, afrag[mt], false, bfrag[nt], (short)0, acc[mt][nt], false, false);
  }

  // ------------------------------------------------------------- epilogue
  const int msub  = (lane >> 4) << 3;        // 0 or 8  (D-matrix M split)
  const int nlane = lane & 15;               // D-matrix N = lane%16
  #pragma unroll
  for (int nt = 0; nt < 4; ++nt) {
    const int n = wn + nt * 16 + nlane;
    const float bval = bias ? bias[n] : 0.0f;
    #pragma unroll
    for (int mt = 0; mt < 2; ++mt) {
      const int mBase = wm + mt * 16 + msub;
      if constexpr (OUTMODE == 1) {          // transposed f16: 8 contig halves
        v8h pk;
        #pragma unroll
        for (int r = 0; r < 8; ++r) {
          float v = acc[mt][nt][r] * scale + bval;
          if (RELU) v = v > 0.f ? v : 0.f;
          pk[r] = (half_t)v;
        }
        *(v8h*)((half_t*)Cv + cOff + (long)n * ldc + mBase) = pk;
      } else {
        #pragma unroll
        for (int r = 0; r < 8; ++r) {
          float v = acc[mt][nt][r] * scale + bval;
          if (RELU) v = v > 0.f ? v : 0.f;
          const long idx = cOff + (long)(mBase + r) * ldc + n;
          if constexpr (OUTMODE == 2) ((float*)Cv)[idx] = v;
          else                        ((half_t*)Cv)[idx] = (half_t)v;
        }
      }
    }
  }
}

// ---------------------------------------------------------------- softmax
// one block (256 thr) per row of 4096 fp32, in place
__global__ __launch_bounds__(256)
void softmax_rows_k(float* __restrict__ attn) {
  float* p = attn + (long)blockIdx.x * NKd;
  const int tid = threadIdx.x, lane = tid & 31, wave = tid >> 5;
  __shared__ float red[8];

  float r[16];
  float m = -3.4e38f;
  #pragma unroll
  for (int i = 0; i < 16; ++i) { r[i] = p[tid + i * 256]; m = fmaxf(m, r[i]); }
  #pragma unroll
  for (int off = 16; off > 0; off >>= 1) m = fmaxf(m, __shfl_xor(m, off, 32));
  if (lane == 0) red[wave] = m;
  __syncthreads();
  if (wave == 0) {
    float t = red[lane & 7];
    #pragma unroll
    for (int off = 4; off > 0; off >>= 1) t = fmaxf(t, __shfl_xor(t, off, 32));
    if (lane == 0) red[0] = t;
  }
  __syncthreads();
  m = red[0];

  float s = 0.f;
  #pragma unroll
  for (int i = 0; i < 16; ++i) { r[i] = __expf(r[i] - m); s += r[i]; }
  #pragma unroll
  for (int off = 16; off > 0; off >>= 1) s += __shfl_xor(s, off, 32);
  __syncthreads();
  if (lane == 0) red[wave] = s;
  __syncthreads();
  if (wave == 0) {
    float t = red[lane & 7];
    #pragma unroll
    for (int off = 4; off > 0; off >>= 1) t += __shfl_xor(t, off, 32);
    if (lane == 0) red[0] = t;
  }
  __syncthreads();
  const float inv = 1.0f / red[0];
  #pragma unroll
  for (int i = 0; i < 16; ++i) p[tid + i * 256] = r[i] * inv;
}

// ---------------------------------------------------------------- launch
extern "C" void kernel_launch(void* const* d_in, const int* in_sizes, int n_in,
                              void* d_out, int out_size, void* d_ws, size_t ws_size,
                              hipStream_t stream) {
  const float* x   = (const float*)d_in[0];
  const float* qin = (const float*)d_in[1];
  const float* W[6]  = {(const float*)d_in[2], (const float*)d_in[4], (const float*)d_in[6],
                        (const float*)d_in[8], (const float*)d_in[10], (const float*)d_in[12]};
  const float* bqv = (const float*)d_in[3];
  const float* bkv = (const float*)d_in[5];
  const float* bvv = (const float*)d_in[7];
  const float* b1  = (const float*)d_in[9];
  const float* b2  = (const float*)d_in[11];
  const float* b3  = (const float*)d_in[13];

  const long M   = (long)BB * NQd;     // 16384 rows for batched projections/MLP
  const long nAct = M * DDd;           // 4,194,304 elems

  char* ws = (char*)d_ws;
  long off = 0;
  auto carve = [&](long bytes) { void* p = ws + off; off += (bytes + 255) & ~255L; return p; };
  half_t* x16   = (half_t*)carve(nAct * 2);
  half_t* q16in = (half_t*)carve(nAct * 2);
  half_t* W16[6];
  for (int i = 0; i < 6; ++i) W16[i] = (half_t*)carve((long)DDd * DDd * 2);
  half_t* q16 = (half_t*)carve(nAct * 2);
  half_t* k16 = (half_t*)carve(nAct * 2);
  half_t* vT  = (half_t*)carve(nAct * 2);   // [256][16384]
  half_t* o16 = (half_t*)carve(nAct * 2);
  half_t* h1  = (half_t*)carve(nAct * 2);
  half_t* h2  = (half_t*)carve(nAct * 2);

  float* h_out = (float*)d_out;
  float* attn  = h_out + nAct;              // 4 x 4096 x 4096 fp32

  dim3 blk(256);
  auto cvtGrid = [](long n) { return dim3((unsigned)((n + 2047) / 2048)); };
  hipLaunchKernelGGL(cvt_f32_f16_k, cvtGrid(nAct), blk, 0, stream, x,   x16,   nAct);
  hipLaunchKernelGGL(cvt_f32_f16_k, cvtGrid(nAct), blk, 0, stream, qin, q16in, nAct);
  for (int i = 0; i < 6; ++i)
    hipLaunchKernelGGL(cvt_f32_f16_k, cvtGrid((long)DDd * DDd), blk, 0, stream,
                       W[i], W16[i], (long)DDd * DDd);

  // ---- projections: [16384,256] x [256,256]^T
  dim3 gProj(DDd / 128, (unsigned)(M / 128), 1);
  hipLaunchKernelGGL((gemm_wmma_k<false, 0, false>), gProj, blk, 0, stream,
                     q16in, W16[0], bqv, q16, DDd, DDd, DDd, DDd, 0L, 0L, 0L, 1.0f);
  hipLaunchKernelGGL((gemm_wmma_k<false, 0, false>), gProj, blk, 0, stream,
                     x16, W16[1], bkv, k16, DDd, DDd, DDd, DDd, 0L, 0L, 0L, 1.0f);
  hipLaunchKernelGGL((gemm_wmma_k<false, 1, false>), gProj, blk, 0, stream,
                     x16, W16[2], bvv, vT, DDd, DDd, DDd, (int)M, 0L, 0L, 0L, 1.0f);

  // ---- scores: per batch  [4096,256] x [4096,256]^T -> fp32 attn, scale 1/16
  dim3 gS(NKd / 128, NQd / 128, BB);
  hipLaunchKernelGGL((gemm_wmma_k<false, 2, false>), gS, blk, 0, stream,
                     q16, k16, (const float*)nullptr, attn,
                     DDd, DDd, DDd, NKd,
                     (long)NQd * DDd, (long)NKd * DDd, (long)NQd * NKd, 0.0625f);

  // ---- softmax in place on attn
  hipLaunchKernelGGL(softmax_rows_k, dim3((unsigned)(BB * NQd)), blk, 0, stream, attn);

  // ---- AV: per batch  attn[4096,4096](f32) x vT -> o16 [4096,256] f16
  dim3 gAV(DDd / 128, NQd / 128, BB);
  hipLaunchKernelGGL((gemm_wmma_k<true, 0, false>), gAV, blk, 0, stream,
                     attn, vT, (const float*)nullptr, o16,
                     NKd, NKd, (int)M, DDd,
                     (long)NQd * NKd, (long)NKd, (long)NQd * DDd, 1.0f);

  // ---- MLP (relu x3), last layer writes fp32 into d_out
  hipLaunchKernelGGL((gemm_wmma_k<false, 0, true>), gProj, blk, 0, stream,
                     o16, W16[3], b1, h1, DDd, DDd, DDd, DDd, 0L, 0L, 0L, 1.0f);
  hipLaunchKernelGGL((gemm_wmma_k<false, 0, true>), gProj, blk, 0, stream,
                     h1, W16[4], b2, h2, DDd, DDd, DDd, DDd, 0L, 0L, 0L, 1.0f);
  hipLaunchKernelGGL((gemm_wmma_k<false, 2, true>), gProj, blk, 0, stream,
                     h2, W16[5], b3, h_out, DDd, DDd, DDd, DDd, 0L, 0L, 0L, 1.0f);
}
